// Linear_84146999263533
// MI455X (gfx1250) — 8-bit matrix operand order repaired and checked by reading, compiled, not run on hardware
//
#include <hip/hip_runtime.h>
#include <hip/hip_bf16.h>


#define M_TOT 16384
#define K_DIM 2048
#define N_DIM 2048

typedef int v8i __attribute__((ext_vector_type(8)));

__global__ __launch_bounds__(256) void quant_rows_kernel(
    const float* __restrict__ x,
    signed char* __restrict__ qa,
    float* __restrict__ a_scale)
{
    const int row = blockIdx.x;
    const int tid = threadIdx.x;

    const float* px = x + (size_t)row * K_DIM + tid * 8;
    const float4 v0 = *(const float4*)(px);
    const float4 v1 = *(const float4*)(px + 4);

    float m = fmaxf(fmaxf(fabsf(v0.x), fabsf(v0.y)),
                    fmaxf(fabsf(v0.z), fabsf(v0.w)));
    m = fmaxf(m, fmaxf(fmaxf(fabsf(v1.x), fabsf(v1.y)),
                       fmaxf(fabsf(v1.z), fabsf(v1.w))));

    __shared__ float red[256];
    red[tid] = m;
    __syncthreads();
    #pragma unroll
    for (int s = 128; s > 0; s >>= 1) {
        if (tid < s) red[tid] = fmaxf(red[tid], red[tid + s]);
        __syncthreads();
    }
    const float mag   = red[0];
    const float scale = (mag == 0.0f) ? 1.0f : mag * (1.0f / 127.0f);
    const float inv   = 1.0f / scale;

    #define Q8(v) ((int)rintf(fminf(fmaxf((v) * inv, -128.0f), 127.0f)))
    const int lo = (Q8(v0.x) & 255)         | ((Q8(v0.y) & 255) << 8) |
                   ((Q8(v0.z) & 255) << 16) | ((Q8(v0.w) & 255) << 24);
    const int hi = (Q8(v1.x) & 255)         | ((Q8(v1.y) & 255) << 8) |
                   ((Q8(v1.z) & 255) << 16) | ((Q8(v1.w) & 255) << 24);
    #undef Q8

    ((int2*)qa)[(((size_t)row * K_DIM) >> 3) + tid] = make_int2(lo, hi);
    if (tid == 0) a_scale[row] = scale;
}

__device__ __forceinline__ v8i load_a_frag(const signed char* __restrict__ base,
                                           unsigned off) {
    const int2 t0 = *(const int2*)(base + off +  0);
    const int2 t1 = *(const int2*)(base + off + 16);
    const int2 t2 = *(const int2*)(base + off + 32);
    const int2 t3 = *(const int2*)(base + off + 48);
    v8i a;
    a[0] = t0.x; a[1] = t0.y; a[2] = t1.x; a[3] = t1.y;
    a[4] = t2.x; a[5] = t2.y; a[6] = t3.x; a[7] = t3.y;
    return a;
}

__device__ __forceinline__ v8i load_b_frag(const signed char* __restrict__ base,
                                           unsigned off) {
    const int2 u0 = *(const int2*)(base + off +  0);
    const int2 u1 = *(const int2*)(base + off + 16);
    const int2 u2 = *(const int2*)(base + off + 32);
    const int2 u3 = *(const int2*)(base + off + 48);
    v8i b;
    b[0] = u0.x; b[1] = u0.y; b[2] = u1.x; b[3] = u1.y;
    b[4] = u2.x; b[5] = u2.y; b[6] = u3.x; b[7] = u3.y;
    return b;
}

__global__ __launch_bounds__(256) void w8a8_gemm_kernel(
    const signed char* __restrict__ qa,
    const float* __restrict__ a_scale,
    const signed char* __restrict__ wt,
    const float* __restrict__ w_scale,
    float* __restrict__ out)
{
    const int lane = threadIdx.x & 31;
    const int wave = threadIdx.x >> 5;
    const int wm   = wave >> 2;
    const int wn   = wave & 3;

    const int mbase = blockIdx.y * 64  + wm * 32;
    const int nbase = blockIdx.x * 256 + wn * 64;

    const int l16  = lane & 15;
    const int half = lane >> 4;

    unsigned offA[2];
    offA[0] = (unsigned)((mbase + l16) * K_DIM + half * 8);
    offA[1] = offA[0] + 16u * K_DIM;

    unsigned offB[4];
    #pragma unroll
    for (int c = 0; c < 4; ++c)
        offB[c] = (unsigned)((nbase + c * 16 + l16) * K_DIM + half * 8);

    const v8i vzero = {};
    v8i acc[2][4];
    #pragma unroll
    for (int r = 0; r < 2; ++r)
        #pragma unroll
        for (int c = 0; c < 4; ++c)
            acc[r][c] = vzero;

    v8i a0[2], b0[4], a1[2], b1[4];
    #pragma unroll
    for (int r = 0; r < 2; ++r) a0[r] = load_a_frag(qa, offA[r]);
    #pragma unroll
    for (int c = 0; c < 4; ++c) b0[c] = load_b_frag(wt, offB[c]);

    #pragma unroll 1
    for (int k0 = 0; k0 < K_DIM; k0 += 128) {
        const signed char* bA1 = qa + (k0 + 64);
        const signed char* bB1 = wt + (k0 + 64);
        const int k2 = (k0 + 128 < K_DIM) ? (k0 + 128) : 0;
        const signed char* bA2 = qa + k2;
        const signed char* bB2 = wt + k2;

        #pragma unroll
        for (int r = 0; r < 2; ++r) a1[r] = load_a_frag(bA1, offA[r]);
        #pragma unroll
        for (int c = 0; c < 4; ++c) b1[c] = load_b_frag(bB1, offB[c]);

        #pragma unroll
        for (int r = 0; r < 2; ++r)
            #pragma unroll
            for (int c = 0; c < 4; ++c)
                acc[r][c] = __builtin_amdgcn_wmma_i32_16x16x64_iu8(
                    true, a0[r], true, b0[c], acc[r][c], false, false);

        #pragma unroll
        for (int r = 0; r < 2; ++r) a0[r] = load_a_frag(bA2, offA[r]);
        #pragma unroll
        for (int c = 0; c < 4; ++c) b0[c] = load_b_frag(bB2, offB[c]);

        #pragma unroll
        for (int r = 0; r < 2; ++r)
            #pragma unroll
            for (int c = 0; c < 4; ++c)
                acc[r][c] = __builtin_amdgcn_wmma_i32_16x16x64_iu8(
                    true, a1[r], true, b1[c], acc[r][c], false, false);
    }

    #pragma unroll
    for (int r = 0; r < 2; ++r) {
        float as[8];
        const int mrow0 = mbase + r * 16 + half * 8;
        #pragma unroll
        for (int i = 0; i < 8; ++i)
            as[i] = a_scale[mrow0 + i];

        #pragma unroll
        for (int c = 0; c < 4; ++c) {
            const int n     = nbase + c * 16 + l16;
            const float wsc = w_scale[n];
            #pragma unroll
            for (int i = 0; i < 8; ++i) {
                out[(size_t)(mrow0 + i) * N_DIM + n] =
                    (float)acc[r][c][i] * as[i] * wsc;
            }
        }
    }
}

extern "C" void kernel_launch(void* const* d_in, const int* in_sizes, int n_in,
                              void* d_out, int out_size, void* d_ws, size_t ws_size,
                              hipStream_t stream) {
    (void)in_sizes; (void)n_in; (void)out_size; (void)ws_size;

    const float*       x   = (const float*)d_in[0];
    const signed char* wt  = (const signed char*)d_in[1];
    const float*       wsc = (const float*)d_in[2];
    float*             out = (float*)d_out;

    signed char* qa      = (signed char*)d_ws;
    float*       a_scale = (float*)((char*)d_ws + (size_t)M_TOT * K_DIM);

    quant_rows_kernel<<<M_TOT, 256, 0, stream>>>(x, qa, a_scale);

    dim3 grid(N_DIM / 256, M_TOT / 64);
    w8a8_gemm_kernel<<<grid, 256, 0, stream>>>(qa, a_scale, wt, wsc, out);
}
